// GRU_90795608637565
// MI455X (gfx1250) — compile-verified, matrix operand order checked by reading, not run on hardware
//
#include <hip/hip_runtime.h>
#include <hip/hip_bf16.h>
#include <math.h>

typedef __bf16 bf16;
typedef __attribute__((ext_vector_type(16))) __bf16 v16bf;
typedef __attribute__((ext_vector_type(8)))  float  v8f;
typedef __attribute__((ext_vector_type(4)))  unsigned int v4u;
typedef __attribute__((ext_vector_type(8)))  int v8i;
typedef __attribute__((ext_vector_type(4)))  int v4i;

#define BS    64
#define HDIM  512
#define TS    1024
#define DIN   96
#define NB    32      // persistent GRU workgroups
#define JSL   16      // hidden units owned per block
#define NGATE 3

// ---------------------------------------------------------------------------
// helpers
// ---------------------------------------------------------------------------
__device__ __forceinline__ float sigmoidf_(float x) { return 1.0f / (1.0f + __expf(-x)); }

// Load one 16x32 bf16 WMMA operand fragment from a row-major [rows][ld] array.
// Per CDNA5 ISA layout (16-bit A-matrix 16x32): lane L (m=L%16, hi=L/16) holds
// K = {hi*8 .. hi*8+7, 16+hi*8 .. 16+hi*8+7} -> two contiguous 16B chunks.
__device__ __forceinline__ v16bf load_frag(const bf16* __restrict__ p0, size_t ld,
                                           int row0, int k0) {
  const int lane = threadIdx.x & 31;
  const int m = lane & 15, hi = lane >> 4;
  const bf16* p = p0 + (size_t)(row0 + m) * ld + (size_t)k0 + hi * 8;
  v16bf f;
  ((uint4*)&f)[0] = *(const uint4*)(p);
  ((uint4*)&f)[1] = *(const uint4*)(p + 16);
  return f;
}

__device__ __forceinline__ v8f wmma_bf16(v16bf a, v16bf b, v8f c) {
  // D(16x16,f32) = A(16x32,bf16) x B(32x16,bf16) + C
  return __builtin_amdgcn_wmma_f32_16x16x32_bf16(false, a, false, b, (short)0, c,
                                                 false, false);
}

// Issue one TDM 2-D tile load (bf16 elements) into LDS at byte offset lds_off.
// D# per CDNA5 ISA ch.8: group0 = {flags, lds_addr, global_addr, type=2},
// group1 = {data_size=2B, tensor_dim0/1, tile_dim0/1, dim0 stride}.
__device__ __forceinline__ void tdm_load_2d(unsigned lds_off, const void* gaddr,
                                            unsigned tdim0, unsigned tdim1,
                                            unsigned tile0, unsigned tile1,
                                            unsigned stride0_elems) {
#if defined(__gfx1250__) && __has_builtin(__builtin_amdgcn_tensor_load_to_lds)
  const unsigned long long ga = (unsigned long long)gaddr;
  v4u g0;
  g0.x = 1u;                                          // count=1 (valid user D#)
  g0.y = lds_off;                                     // lds_addr (bytes)
  g0.z = (unsigned)(ga & 0xffffffffu);                // global_addr[31:0]
  g0.w = (unsigned)((ga >> 32) & 0x01ffffffu) | (2u << 30);  // addr[56:32] | type=2
  const unsigned w0 = (1u << 16);                     // data_size=1 -> 2 bytes
  const unsigned w1 = (tdim0 & 0xffffu) << 16;        // tensor_dim0[15:0]
  const unsigned w2 = ((tdim0 >> 16) & 0xffffu) | ((tdim1 & 0xffffu) << 16);
  const unsigned w3 = ((tdim1 >> 16) & 0xffffu) | ((tile0 & 0xffffu) << 16);
  const unsigned w4 = (tile1 & 0xffffu);              // tile_dim1; tile_dim2=0
  const unsigned w5 = stride0_elems;                  // tensor_dim0_stride[31:0]
  v8i g1 = {(int)w0, (int)w1, (int)w2, (int)w3, (int)w4, (int)w5, 0, 0};
  v4i gz = {0, 0, 0, 0};
#if __clang_major__ >= 23
  v8i gz8 = {0, 0, 0, 0, 0, 0, 0, 0};
  __builtin_amdgcn_tensor_load_to_lds(g0, g1, gz, gz, gz8, 0);
#else
  __builtin_amdgcn_tensor_load_to_lds(g0, g1, gz, gz, 0);
#endif
#else
  (void)lds_off; (void)gaddr; (void)tdim0; (void)tdim1;
  (void)tile0; (void)tile1; (void)stride0_elems;
#endif
}

// Grid-wide sense-reversing barrier for the persistent GRU kernel.
__device__ __forceinline__ void grid_sync(unsigned* bar) {
  __syncthreads();
  if (threadIdx.x == 0) {
    volatile unsigned* vgen = bar + 1;
    const unsigned gen = *vgen;
    __threadfence();
    if (atomicAdd(bar, 1u) == NB - 1) {
      atomicExch(bar, 0u);
      __threadfence();
      atomicAdd(bar + 1, 1u);
    } else {
      while (*vgen == gen) { __builtin_amdgcn_s_sleep(1); }
    }
    __threadfence();
  }
  __syncthreads();
}

// ---------------------------------------------------------------------------
// small utility kernels
// ---------------------------------------------------------------------------
__global__ void zero_kernel(unsigned* p, size_t n) {
  size_t i = (size_t)blockIdx.x * blockDim.x + threadIdx.x;
  if (i < n) p[i] = 0u;
}

__global__ void cvt_kernel(const float* __restrict__ s, bf16* __restrict__ d, size_t n) {
  size_t i = (size_t)blockIdx.x * blockDim.x + threadIdx.x;
  if (i < n) d[i] = (bf16)s[i];
}

// ---------------------------------------------------------------------------
// Row-block GEMM + LeakyReLU + LayerNorm (used for encoder and fc1).
// Each block: 16 rows x 512 cols, full row held in LDS so LN can fuse.
// k-loop is 2x-unrolled ping-pong pipelined: two fragment buffer sets swap
// roles each half-iteration, so the next tile's loads overlap the current
// WMMAs with zero register-rotation copies.
// ---------------------------------------------------------------------------
__global__ __launch_bounds__(256, 2)
void rowblock_gemm_ln(const bf16* __restrict__ A, int K,
                      const bf16* __restrict__ W,
                      const float* __restrict__ bias,
                      const float* __restrict__ gamma,
                      const float* __restrict__ beta,
                      bf16* __restrict__ out, int permute_bt) {
  __shared__ float sbuf[16][HDIM];
  __shared__ float red[16][16][2];
  __shared__ float mv[16][2];

  const int row0 = blockIdx.x * 16;
  const int wave = threadIdx.x >> 5;
  const int lane = threadIdx.x & 31;
  const int n = lane & 15, hi = lane >> 4;

  v8f acc[4] = {};
  v16bf a0 = load_frag(A, (size_t)K, row0, 0);
  v16bf b0[4];
#pragma unroll
  for (int q = 0; q < 4; ++q) b0[q] = load_frag(W, (size_t)K, (wave + 8 * q) * 16, 0);

  int kk = 0;
  while (kk + 64 <= K) {
    // stage 1: load buf1 @ kk+32, compute with buf0 @ kk
    v16bf a1 = load_frag(A, (size_t)K, row0, kk + 32);
    v16bf b1[4];
#pragma unroll
    for (int q = 0; q < 4; ++q)
      b1[q] = load_frag(W, (size_t)K, (wave + 8 * q) * 16, kk + 32);
#pragma unroll
    for (int q = 0; q < 4; ++q) acc[q] = wmma_bf16(a0, b0[q], acc[q]);
    // stage 2: load buf0 @ kk+64 (clamped, benign at tail), compute with buf1
    const int k2 = (kk + 64 < K) ? kk + 64 : 0;
    a0 = load_frag(A, (size_t)K, row0, k2);
#pragma unroll
    for (int q = 0; q < 4; ++q) b0[q] = load_frag(W, (size_t)K, (wave + 8 * q) * 16, k2);
#pragma unroll
    for (int q = 0; q < 4; ++q) acc[q] = wmma_bf16(a1, b1[q], acc[q]);
    kk += 64;
  }
  if (kk < K) {  // odd tail (e.g. encoder K=96): buf0 already holds frags @ kk
#pragma unroll
    for (int q = 0; q < 4; ++q) acc[q] = wmma_bf16(a0, b0[q], acc[q]);
  }

#pragma unroll
  for (int q = 0; q < 4; ++q) {
    const int col = (wave + 8 * q) * 16 + n;
#pragma unroll
    for (int rr = 0; rr < 8; ++rr) {
      const int m = rr + 8 * hi;            // C layout: VGPR rr, lane -> M = rr + 8*hi
      float v = acc[q][rr] + bias[col];
      sbuf[m][col] = v >= 0.0f ? v : 0.01f * v;   // LeakyReLU
    }
  }
  __syncthreads();

  // LayerNorm over the 512-wide rows: 16 threads per row.
  const int row = threadIdx.x >> 4, part = threadIdx.x & 15;
  float s = 0.0f, ss = 0.0f;
  for (int c = part; c < HDIM; c += 16) { float v = sbuf[row][c]; s += v; ss += v * v; }
  red[row][part][0] = s; red[row][part][1] = ss;
  __syncthreads();
  if (part == 0) {
    float S = 0.0f, SS = 0.0f;
    for (int i = 0; i < 16; ++i) { S += red[row][i][0]; SS += red[row][i][1]; }
    const float mean = S * (1.0f / HDIM);
    const float var = SS * (1.0f / HDIM) - mean * mean;
    mv[row][0] = mean; mv[row][1] = rsqrtf(var + 1e-5f);
  }
  __syncthreads();

  const int gr = row0 + row;
  const size_t orow = permute_bt ? ((size_t)(gr & (TS - 1)) * BS + (size_t)(gr >> 10))
                                 : (size_t)gr;
  bf16* op = out + orow * HDIM;
  for (int c = part; c < HDIM; c += 16) {
    float v = (sbuf[row][c] - mv[row][0]) * mv[row][1] * gamma[c] + beta[c];
    op[c] = (bf16)v;
  }
}

// ---------------------------------------------------------------------------
// Persistent GRU: one WMMA step for one layer.
// Block owns JSL hidden units (cols j0..j0+15) for each of the 3 gates.
// Wave schedule (balanced, 3 WMMA per wave per k-step):
//   wave = (mt, half); mt = M-tile of the 64-row batch, 2 waves per mt.
//   half 0: acc_a = r gate (input+recurrent merged), acc_b = i_n
//   half 1: acc_a = z gate (merged),                 acc_b = h_n
// (r and z only ever need i+h summed; only gate n needs the two parts apart.)
// k-loop: 2x-unrolled ping-pong pipeline over the 5 fragments
// (2 global A, 3 LDS B) -> loads overlap WMMA, no rotation copies.
// ---------------------------------------------------------------------------
__device__ __forceinline__ void gru_layer(
    const bf16* __restrict__ xin,        // [BS][H] bf16 (time-major slice)
    const bf16* __restrict__ hin_b,      // [BS][H] bf16
    const float* __restrict__ hin_f,     // [BS][H] f32
    const bf16* __restrict__ s_wih, const bf16* __restrict__ s_whh,  // LDS [3][JSL][H]
    const float* __restrict__ bih, const float* __restrict__ bhh,    // [3H]
    bf16* __restrict__ hout_b, float* __restrict__ hout_f,
    bf16* __restrict__ tops_row,         // nullptr for layer 0
    float* s_r, float* s_z, float* s_in, float* s_hn, int j0) {
  const int wave = threadIdx.x >> 5;
  const int lane = threadIdx.x & 31;
  const int n = lane & 15, hi = lane >> 4;
  const int mt = wave & 3, half = wave >> 2;

  const bf16* wA = s_wih + (half ? (size_t)1 * JSL * HDIM : 0);   // r or z (input)
  const bf16* wB = s_whh + (half ? (size_t)1 * JSL * HDIM : 0);   // r or z (recurrent)
  const bf16* wN = half ? (s_whh + (size_t)2 * JSL * HDIM)        // h_n weights
                        : (s_wih + (size_t)2 * JSL * HDIM);       // i_n weights

  v8f acc_a = {}, acc_b = {};
  v16bf ax0 = load_frag(xin,   HDIM, mt * 16, 0);
  v16bf ah0 = load_frag(hin_b, HDIM, mt * 16, 0);
  v16bf bA0 = load_frag(wA, HDIM, 0, 0);
  v16bf bB0 = load_frag(wB, HDIM, 0, 0);
  v16bf bN0 = load_frag(wN, HDIM, 0, 0);

  for (int kk = 0; kk < HDIM; kk += 64) {   // HDIM is a multiple of 64
    // stage 1: load buf1 @ kk+32, compute with buf0 @ kk
    v16bf ax1 = load_frag(xin,   HDIM, mt * 16, kk + 32);
    v16bf ah1 = load_frag(hin_b, HDIM, mt * 16, kk + 32);
    v16bf bA1 = load_frag(wA, HDIM, 0, kk + 32);
    v16bf bB1 = load_frag(wB, HDIM, 0, kk + 32);
    v16bf bN1 = load_frag(wN, HDIM, 0, kk + 32);
    acc_a = wmma_bf16(ax0, bA0, acc_a);
    acc_a = wmma_bf16(ah0, bB0, acc_a);
    acc_b = wmma_bf16(half ? ah0 : ax0, bN0, acc_b);
    // stage 2: load buf0 @ kk+64 (clamped at tail, benign), compute with buf1
    const int k2 = (kk + 64 < HDIM) ? kk + 64 : 0;
    ax0 = load_frag(xin,   HDIM, mt * 16, k2);
    ah0 = load_frag(hin_b, HDIM, mt * 16, k2);
    bA0 = load_frag(wA, HDIM, 0, k2);
    bB0 = load_frag(wB, HDIM, 0, k2);
    bN0 = load_frag(wN, HDIM, 0, k2);
    acc_a = wmma_bf16(ax1, bA1, acc_a);
    acc_a = wmma_bf16(ah1, bB1, acc_a);
    acc_b = wmma_bf16(half ? ah1 : ax1, bN1, acc_b);
  }

#pragma unroll
  for (int rr = 0; rr < 8; ++rr) {
    const int m = mt * 16 + rr + 8 * hi;   // batch row
    const int e = m * JSL + n;             // n = owned-unit index
    if (half == 0) { s_r[e] = acc_a[rr]; s_in[e] = acc_b[rr]; }
    else           { s_z[e] = acc_a[rr]; s_hn[e] = acc_b[rr]; }
  }
  __syncthreads();

  // Gate epilogue over BS*JSL = 1024 elements.
  for (int e = threadIdx.x; e < BS * JSL; e += blockDim.x) {
    const int m = e / JSL, jj = e - m * JSL;
    const int j = j0 + jj;
    const float r = sigmoidf_(s_r[e] + bih[j] + bhh[j]);
    const float z = sigmoidf_(s_z[e] + bih[HDIM + j] + bhh[HDIM + j]);
    const float nn = tanhf(s_in[e] + bih[2 * HDIM + j] + r * (s_hn[e] + bhh[2 * HDIM + j]));
    const float hprev = hin_f[(size_t)m * HDIM + j];
    const float hnew = (1.0f - z) * nn + z * hprev;
    hout_f[(size_t)m * HDIM + j] = hnew;
    hout_b[(size_t)m * HDIM + j] = (bf16)hnew;
    if (tops_row) tops_row[(size_t)m * HDIM + j] = (bf16)hnew;
  }
}

__global__ __launch_bounds__(256, 1)
void gru_persistent(const bf16* __restrict__ emb,                       // [TS][BS][H]
                    const bf16* __restrict__ wih0, const bf16* __restrict__ whh0,
                    const bf16* __restrict__ wih1, const bf16* __restrict__ whh1,
                    const float* __restrict__ bih0, const float* __restrict__ bhh0,
                    const float* __restrict__ bih1, const float* __restrict__ bhh1,
                    bf16* __restrict__ h0b, bf16* __restrict__ h1b,     // [2][BS][H]
                    float* __restrict__ h0f, float* __restrict__ h1f,   // [2][BS][H]
                    bf16* __restrict__ tops,                            // [TS][BS][H]
                    unsigned* __restrict__ bar) {
  extern __shared__ __align__(16) char smem[];
  const unsigned WSLICE = NGATE * JSL * HDIM;         // 24576 elems = 48 KB
  bf16* s_wih0 = (bf16*)smem;
  bf16* s_whh0 = s_wih0 + WSLICE;
  bf16* s_wih1 = s_whh0 + WSLICE;
  bf16* s_whh1 = s_wih1 + WSLICE;                     // weights total: 192 KB LDS
  float* s_r  = (float*)(s_whh1 + WSLICE);
  float* s_z  = s_r  + BS * JSL;
  float* s_in = s_z  + BS * JSL;
  float* s_hn = s_in + BS * JSL;                      // gate scratch: 16 KB

  const int j0 = blockIdx.x * JSL;

  // --- One-time weight staging: Tensor Data Mover, one 2-D tile per gate ---
  // (16 rows x 512 cols bf16, row stride 512 elems). Wave 0 issues 12 D#s.
  if (threadIdx.x == 0) {
    const bf16* srcs[4] = {wih0, whh0, wih1, whh1};
#pragma unroll
    for (int wsel = 0; wsel < 4; ++wsel) {
#pragma unroll
      for (int g = 0; g < NGATE; ++g) {
        const unsigned lds_off = (unsigned)(wsel * WSLICE + g * JSL * HDIM) * 2u;
        tdm_load_2d(lds_off, srcs[wsel] + (size_t)(g * HDIM + j0) * HDIM,
                    HDIM, NGATE * HDIM, HDIM, JSL, HDIM);
      }
    }
#if defined(__gfx1250__) && __has_builtin(__builtin_amdgcn_s_wait_tensorcnt)
    __builtin_amdgcn_s_wait_tensorcnt(0);
#endif
  }
  __syncthreads();
  // Authoritative cooperative copy (same bytes; covers any D# field skew).
  for (int idx = threadIdx.x; idx < (int)WSLICE; idx += blockDim.x) {
    const int row = idx / HDIM, k = idx - row * HDIM;
    const int g = row / JSL, jj = row - g * JSL;
    const size_t src = (size_t)(g * HDIM + j0 + jj) * HDIM + k;
    s_wih0[idx] = wih0[src];
    s_whh0[idx] = whh0[src];
    s_wih1[idx] = wih1[src];
    s_whh1[idx] = whh1[src];
  }
  __syncthreads();

  const size_t SZ = (size_t)BS * HDIM;
  int p = 0;
  for (int t = 0; t < TS; ++t) {
    if (t + 2 < TS) {  // pull the t+2 timestep toward L2/L0 (global_prefetch_b8)
      const char* pf = (const char*)(emb + (size_t)(t + 2) * SZ);
      __builtin_prefetch(pf + (size_t)threadIdx.x * 256, 0, 0);
    }
    const bf16* xt = emb + (size_t)t * SZ;
    // layer 0: reads h0[p], writes h0[p^1]
    gru_layer(xt, h0b + p * SZ, h0f + p * SZ, s_wih0, s_whh0, bih0, bhh0,
              h0b + (p ^ 1) * SZ, h0f + (p ^ 1) * SZ, nullptr,
              s_r, s_z, s_in, s_hn, j0);
    grid_sync(bar);
    // layer 1: input = new h0, reads h1[p], writes h1[p^1] and tops[t]
    gru_layer(h0b + (p ^ 1) * SZ, h1b + p * SZ, h1f + p * SZ, s_wih1, s_whh1,
              bih1, bhh1, h1b + (p ^ 1) * SZ, h1f + (p ^ 1) * SZ,
              tops + (size_t)t * SZ, s_r, s_z, s_in, s_hn, j0);
    grid_sync(bar);
    p ^= 1;
  }
  // t=1023 writes buffer 0 -> final states live at h0f/h1f + 0.
}

// ---------------------------------------------------------------------------
// Tail kernels (negligible FLOPs)
// ---------------------------------------------------------------------------
__global__ void fc2_kernel(const bf16* __restrict__ hfc,   // [TS*BS][H], row = t*BS+b
                           const float* __restrict__ W,    // [2][H]
                           const float* __restrict__ bias, // [2]
                           float* __restrict__ out) {      // [BS][TS][2]
  const int r = blockIdx.x * blockDim.x + threadIdx.x;
  if (r >= TS * BS) return;
  const int t = r / BS, b = r - t * BS;
  float a0 = bias[0], a1 = bias[1];
  const bf16* hr = hfc + (size_t)r * HDIM;
  for (int k = 0; k < HDIM; ++k) {
    const float h = (float)hr[k];
    a0 += h * W[k];
    a1 += h * W[HDIM + k];
  }
  const size_t o = ((size_t)b * TS + t) * 2;
  out[o] = a0; out[o + 1] = a1;
}

__global__ void cls1_kernel(const float* __restrict__ h0f, const float* __restrict__ h1f,
                            const float* __restrict__ Wc1, const float* __restrict__ bc1,
                            const float* __restrict__ gbn, const float* __restrict__ bebn,
                            float* __restrict__ cbuf) {    // [BS][H]
  const int b = blockIdx.x;
  const float* h0 = h0f + (size_t)b * HDIM;
  const float* h1 = h1f + (size_t)b * HDIM;
  for (int j = threadIdx.x; j < HDIM; j += blockDim.x) {
    const float* w = Wc1 + (size_t)j * (2 * HDIM);
    float acc = bc1[j];
    for (int k = 0; k < HDIM; ++k) acc += h0[k] * w[k];
    for (int k = 0; k < HDIM; ++k) acc += h1[k] * w[HDIM + k];
    acc = acc >= 0.0f ? acc : 0.01f * acc;                  // LeakyReLU
    acc = acc * rsqrtf(1.0f + 1e-5f) * gbn[j] + bebn[j];    // BatchNorm (eval, default stats)
    cbuf[(size_t)b * HDIM + j] = acc;
  }
}

__global__ void cls2_kernel(const float* __restrict__ cbuf,
                            const float* __restrict__ W,    // [9][H]
                            const float* __restrict__ bias, // [9]
                            float* __restrict__ out) {      // [BS][9]
  const int idx = blockIdx.x * blockDim.x + threadIdx.x;
  if (idx >= BS * 9) return;
  const int b = idx / 9, o = idx - b * 9;
  float acc = bias[o];
  const float* c = cbuf + (size_t)b * HDIM;
  const float* w = W + (size_t)o * HDIM;
  for (int k = 0; k < HDIM; ++k) acc += c[k] * w[k];
  out[idx] = acc;
}

// ---------------------------------------------------------------------------
// host launcher
// ---------------------------------------------------------------------------
extern "C" void kernel_launch(void* const* d_in, const int* in_sizes, int n_in,
                              void* d_out, int out_size, void* d_ws, size_t ws_size,
                              hipStream_t stream) {
  (void)in_sizes; (void)n_in; (void)out_size; (void)ws_size;
  const float* x      = (const float*)d_in[0];
  const float* W_enc  = (const float*)d_in[1];
  const float* b_enc  = (const float*)d_in[2];
  const float* g_enc  = (const float*)d_in[3];
  const float* be_enc = (const float*)d_in[4];
  const float* W_ih0  = (const float*)d_in[5];
  const float* W_hh0  = (const float*)d_in[6];
  const float* b_ih0  = (const float*)d_in[7];
  const float* b_hh0  = (const float*)d_in[8];
  const float* W_ih1  = (const float*)d_in[9];
  const float* W_hh1  = (const float*)d_in[10];
  const float* b_ih1  = (const float*)d_in[11];
  const float* b_hh1  = (const float*)d_in[12];
  const float* W_fc1  = (const float*)d_in[13];
  const float* b_fc1  = (const float*)d_in[14];
  const float* g_fc   = (const float*)d_in[15];
  const float* be_fc  = (const float*)d_in[16];
  const float* W_fc2  = (const float*)d_in[17];
  const float* b_fc2  = (const float*)d_in[18];
  const float* W_cls1 = (const float*)d_in[19];
  const float* b_cls1 = (const float*)d_in[20];
  const float* g_bn   = (const float*)d_in[21];
  const float* be_bn  = (const float*)d_in[22];
  const float* W_cls2 = (const float*)d_in[23];
  const float* b_cls2 = (const float*)d_in[24];
  float* out = (float*)d_out;

  // workspace carve-up
  char* ws = (char*)d_ws;
  size_t off = 0;
  auto alloc = [&](size_t bytes) -> void* {
    void* p = ws + off;
    off = (off + bytes + 255) & ~(size_t)255;
    return p;
  };
  const size_t NROWS = (size_t)BS * TS;                 // 65536
  bf16* x_bf     = (bf16*)alloc(NROWS * DIN * 2);       // [b*TS+t][96]
  bf16* wenc_bf  = (bf16*)alloc((size_t)HDIM * DIN * 2);
  bf16* wih0_bf  = (bf16*)alloc((size_t)3 * HDIM * HDIM * 2);
  bf16* whh0_bf  = (bf16*)alloc((size_t)3 * HDIM * HDIM * 2);
  bf16* wih1_bf  = (bf16*)alloc((size_t)3 * HDIM * HDIM * 2);
  bf16* whh1_bf  = (bf16*)alloc((size_t)3 * HDIM * HDIM * 2);
  bf16* wfc1_bf  = (bf16*)alloc((size_t)HDIM * HDIM * 2);
  bf16* emb_bf   = (bf16*)alloc((size_t)TS * BS * HDIM * 2);   // time-major
  bf16* tops_bf  = (bf16*)alloc((size_t)TS * BS * HDIM * 2);   // time-major
  bf16* hfc_bf   = (bf16*)alloc(NROWS * HDIM * 2);
  // contiguous state block (zeroed every launch)
  char* state    = (char*)alloc(131072 + 131072 + 262144 + 262144 + 131072 + 256);
  bf16*  h0b  = (bf16*)(state);                      // [2][BS][H] bf16
  bf16*  h1b  = (bf16*)(state + 131072);
  float* h0f  = (float*)(state + 262144);            // [2][BS][H] f32
  float* h1f  = (float*)(state + 524288);
  float* cbuf = (float*)(state + 786432);            // [BS][H]
  unsigned* bar = (unsigned*)(state + 917504);
  const size_t state_words = 917760 / 4;

  // 0) zero recurrent state + barrier
  zero_kernel<<<(unsigned)((state_words + 255) / 256), 256, 0, stream>>>(
      (unsigned*)state, state_words);

  // 1) f32 -> bf16 conversions (weights once per launch + activations)
  auto cvt = [&](const float* s, bf16* d, size_t n) {
    cvt_kernel<<<(unsigned)((n + 255) / 256), 256, 0, stream>>>(s, d, n);
  };
  cvt(x,     x_bf,    NROWS * DIN);
  cvt(W_enc, wenc_bf, (size_t)HDIM * DIN);
  cvt(W_ih0, wih0_bf, (size_t)3 * HDIM * HDIM);
  cvt(W_hh0, whh0_bf, (size_t)3 * HDIM * HDIM);
  cvt(W_ih1, wih1_bf, (size_t)3 * HDIM * HDIM);
  cvt(W_hh1, whh1_bf, (size_t)3 * HDIM * HDIM);
  cvt(W_fc1, wfc1_bf, (size_t)HDIM * HDIM);

  // 2) encoder: emb = LN(leaky(x @ W_enc^T + b)); written time-major bf16
  rowblock_gemm_ln<<<(unsigned)(NROWS / 16), 256, 0, stream>>>(
      x_bf, DIN, wenc_bf, b_enc, g_enc, be_enc, emb_bf, /*permute_bt=*/1);

  // 3) persistent 2-layer GRU (weights LDS-resident via TDM, 2 grid barriers/step)
  const int DYNLDS = 4 * NGATE * JSL * HDIM * 2 + 4 * BS * JSL * 4;  // 212992 B
  (void)hipFuncSetAttribute((const void*)gru_persistent,
                            hipFuncAttributeMaxDynamicSharedMemorySize, DYNLDS);
  gru_persistent<<<NB, 256, DYNLDS, stream>>>(
      emb_bf, wih0_bf, whh0_bf, wih1_bf, whh1_bf,
      b_ih0, b_hh0, b_ih1, b_hh1,
      h0b, h1b, h0f, h1f, tops_bf, bar);

  // 4) fc head: hfc = LN(leaky(tops @ W_fc1^T + b)); then 512->2 projection
  rowblock_gemm_ln<<<(unsigned)(NROWS / 16), 256, 0, stream>>>(
      tops_bf, HDIM, wfc1_bf, b_fc1, g_fc, be_fc, hfc_bf, /*permute_bt=*/0);
  fc2_kernel<<<(unsigned)(NROWS / 256), 256, 0, stream>>>(hfc_bf, W_fc2, b_fc2, out);

  // 5) cls head on final hidden states (buffer 0 after 1024 steps)
  cls1_kernel<<<BS, 256, 0, stream>>>(h0f, h1f, W_cls1, b_cls1, g_bn, be_bn, cbuf);
  cls2_kernel<<<3, 256, 0, stream>>>(cbuf, W_cls2, b_cls2, out + (size_t)BS * TS * 2);
}
